// GcisApBasicBlock_13572096655694
// MI455X (gfx1250) — matrix operand order repaired and checked by reading, not compiled, not run on hardware
//
#include <hip/hip_runtime.h>

// ---------------- problem constants ----------------
#define CIN   32
#define COUT  64
#define DIN   48
#define DOUT  24
#define NPOS  (DOUT*DOUT*DOUT)   // 13824 output spatial positions
#define NT    128                // N-tile per block (13824 % 128 == 0 -> 108 blocks)
#define BKP   40                 // 32 K + 8 pad bf16 -> 80B rows (16B aligned)
#define EPS_F 1e-5f

typedef __attribute__((ext_vector_type(16))) __bf16 v16bf;
typedef __attribute__((ext_vector_type(8)))  float  v8f;

static __device__ __forceinline__ __bf16 f2bf(float f) { return (__bf16)f; }

// A fragment for 16-bit A 16x32 WMMA layout (ISA 7.12.2):
// lanes 0-15 (kg=0) hold K {0..7,16..23}; lanes 16-31 (kg=1) hold K {8..15,24..31};
// row = 32 contiguous bf16 in LDS -> two 16B ds_load_b128 per fragment.
static __device__ __forceinline__ v16bf load_afrag(const __bf16* row, int kg) {
  v16bf a;
  const __bf16* p0 = row + 8 * kg;
#pragma unroll
  for (int e = 0; e < 8; ++e) a[e] = p0[e];
  const __bf16* p1 = row + 16 + 8 * kg;
#pragma unroll
  for (int e = 0; e < 8; ++e) a[8 + e] = p1[e];
  return a;
}

// B fragment: same K order as A: e<8 -> K=8*kg+e, e>=8 -> K=16+8*kg+(e-8); Bt stored [N][K].
static __device__ __forceinline__ v16bf load_bfrag(const __bf16* brow, int kg) {
  v16bf b;
  const __bf16* p = brow + 8 * kg;
#pragma unroll
  for (int e = 0; e < 8; ++e) b[e] = p[e];
  const __bf16* p1 = brow + 16 + 8 * kg;
#pragma unroll
  for (int e = 0; e < 8; ++e) b[8 + e] = p1[e];
  return b;
}

// ---------------- router: logits -> argmax -> sel[b] ----------------
__global__ __launch_bounds__(256)
void k_router(const float* __restrict__ prompt, const float* __restrict__ rw,
              const float* __restrict__ rb, int* __restrict__ sel) {
  __shared__ float logits[8];
  int tid = threadIdx.x;
  int g = tid >> 5, lane = tid & 31;   // 8 wave32 groups
  int b = g >> 2, e = g & 3;           // (batch, expert)
  float s = 0.f;
  for (int i = lane; i < 512; i += 32) s += prompt[b * 512 + i] * rw[e * 512 + i];
  for (int off = 16; off > 0; off >>= 1) s += __shfl_down(s, off, 32);
  if (lane == 0) logits[g] = s + rb[e];
  __syncthreads();
  if (tid < 2) {
    int best = 0; float bv = logits[tid * 4];
    for (int e2 = 1; e2 < 4; ++e2) {
      float v = logits[tid * 4 + e2];
      if (v > bv) { bv = v; best = e2; }   // strict > : first max (matches argmax)
    }
    sel[tid] = best;
  }
}

// ---------------- weight repack: [e][o][c][27] f32 -> [e][t][o][c] bf16 ----------------
__global__ __launch_bounds__(256)
void k_prep_w1(const float* __restrict__ w1, __bf16* __restrict__ w1t) {
  int idx = blockIdx.x * 256 + threadIdx.x;
  if (idx >= 4 * 27 * 64 * 32) return;
  int c = idx & 31, o = (idx >> 5) & 63, t = (idx / 2048) % 27, e = idx / (2048 * 27);
  w1t[idx] = f2bf(w1[((e * 64 + o) * 32 + c) * 27 + t]);
}

// conv2 weights -> [e][ks=2t+half][o][c32] bf16
__global__ __launch_bounds__(256)
void k_prep_w2(const float* __restrict__ w2, __bf16* __restrict__ w2t) {
  int idx = blockIdx.x * 256 + threadIdx.x;
  if (idx >= 4 * 54 * 64 * 32) return;
  int cc = idx & 31, o = (idx >> 5) & 63, ks = (idx / 2048) % 54, e = idx / (2048 * 54);
  int t = ks >> 1, c = (ks & 1) * 32 + cc;
  w2t[idx] = f2bf(w2[((e * 64 + o) * 64 + c) * 27 + t]);
}

// ---------------- conv1 (3x3x3 s2 p1, 32->64) + fused conv3 (1x1x1 s2) ----------------
__global__ __launch_bounds__(256)
void k_conv1_conv3(const float* __restrict__ x, const __bf16* __restrict__ w1t,
                   const float* __restrict__ w3, const int* __restrict__ sel,
                   float* __restrict__ y1, float* __restrict__ y3) {
  __shared__ __align__(16) __bf16 At[2][COUT][32];   // M x K32, double buffered
  __shared__ __align__(16) __bf16 Bt[2][NT][BKP];    // N x K32 (transposed), padded rows
  __shared__ __align__(16) __bf16 A3[COUT][32];      // conv3 weights

  const int b   = blockIdx.y;
  const int n0  = blockIdx.x * NT;
  const int tid = threadIdx.x;
  const int s   = sel[b];

  const float* xb = x + (size_t)b * CIN * DIN * DIN * DIN;

  // conv3 weights: contiguous [64][32] -> LDS once
  {
    const float* w3e = w3 + (size_t)s * COUT * CIN;
    for (int i = tid; i < COUT * 32; i += 256) A3[i >> 5][i & 31] = f2bf(w3e[i]);
  }

  const int wid  = tid >> 5, lane = tid & 31;
  const int nsub = wid * 16;
  const int col  = lane & 15, kg = lane >> 4;

  auto stageA = [&](int t, int buf) {
    // one 16B vector copy per thread (2048 bf16 total)
    const uint4* src = (const uint4*)(w1t + (size_t)(s * 27 + t) * 2048);
    ((uint4*)&At[buf][0][0])[tid] = src[tid];
  };
  auto stageB = [&](int t, int buf) {
    const int dz = t / 9 - 1, dy = (t / 3) % 3 - 1, dx = t % 3 - 1;
#pragma unroll
    for (int r = 0; r < 16; ++r) {
      int i = r * 256 + tid;
      int c = i / NT, j = i % NT;           // consecutive tids -> consecutive j (coalesced x)
      int p = n0 + j;
      int d = p / (DOUT * DOUT), hh = (p / DOUT) % DOUT, ww = p % DOUT;
      int iz = 2 * d + dz, iy = 2 * hh + dy, ix = 2 * ww + dx;
      float v = 0.f;
      if ((unsigned)iz < (unsigned)DIN && (unsigned)iy < (unsigned)DIN &&
          (unsigned)ix < (unsigned)DIN)
        v = xb[((c * DIN + iz) * DIN + iy) * DIN + ix];
      Bt[buf][j][c] = f2bf(v);
    }
  };

  v8f acc[4] = {};

  stageA(0, 0);
  stageB(0, 0);
  for (int t = 0; t < 27; ++t) {           // single barrier per k-step, ping-pong buffers
    const int buf = t & 1;
    __syncthreads();
    if (t + 1 < 27) { stageA(t + 1, buf ^ 1); stageB(t + 1, buf ^ 1); }
    v16bf bf = load_bfrag(&Bt[buf][nsub + col][0], kg);
#pragma unroll
    for (int mt = 0; mt < 4; ++mt) {
      v16bf af = load_afrag(&At[buf][mt * 16 + col][0], kg);
      acc[mt] = __builtin_amdgcn_wmma_f32_16x16x32_bf16(
          false, af, false, bf, (short)0, acc[mt], false, false);
    }
  }

  // conv3: its im2col column == conv1 center tap (t=13); re-stage once, no branches in hot loop
  __syncthreads();
  stageB(13, 0);
  __syncthreads();
  v8f acc3[4] = {};
  {
    v16bf bf = load_bfrag(&Bt[0][nsub + col][0], kg);
#pragma unroll
    for (int mt = 0; mt < 4; ++mt) {
      v16bf af = load_afrag(&A3[mt * 16 + col][0], kg);
      acc3[mt] = __builtin_amdgcn_wmma_f32_16x16x32_bf16(
          false, af, false, bf, (short)0, acc3[mt], false, false);
    }
  }

  float* y1b = y1 + (size_t)b * COUT * NPOS;
  float* y3b = y3 + (size_t)b * COUT * NPOS;
  const int n = n0 + nsub + col;
#pragma unroll
  for (int mt = 0; mt < 4; ++mt)
#pragma unroll
    for (int r = 0; r < 8; ++r) {
      int m = mt * 16 + r + 8 * kg;   // C/D layout: vgpr r -> row r (+8 for lanes 16..31)
      y1b[(size_t)m * NPOS + n] = acc[mt][r];
      y3b[(size_t)m * NPOS + n] = acc3[mt][r];
    }
}

// ---------------- conv2 (3x3x3 s1 p1, 64->64) on bf16 activations ----------------
__global__ __launch_bounds__(256)
void k_conv2(const __bf16* __restrict__ a1, const __bf16* __restrict__ w2t,
             const int* __restrict__ sel, float* __restrict__ y2) {
  __shared__ __align__(16) __bf16 At[2][COUT][32];
  __shared__ __align__(16) __bf16 Bt[2][NT][BKP];

  const int b   = blockIdx.y;
  const int n0  = blockIdx.x * NT;
  const int tid = threadIdx.x;
  const int s   = sel[b];

  const __bf16* ab = a1 + (size_t)b * COUT * NPOS;

  const int wid  = tid >> 5, lane = tid & 31;
  const int nsub = wid * 16;
  const int col  = lane & 15, kg = lane >> 4;

  auto stageA = [&](int ks, int buf) {
    const uint4* src = (const uint4*)(w2t + (size_t)(s * 54 + ks) * 2048);
    ((uint4*)&At[buf][0][0])[tid] = src[tid];
  };
  auto stageB = [&](int ks, int buf) {
    const int t = ks >> 1, ch = (ks & 1) * 32;
    const int dz = t / 9 - 1, dy = (t / 3) % 3 - 1, dx = t % 3 - 1;
#pragma unroll
    for (int r = 0; r < 16; ++r) {
      int i = r * 256 + tid;
      int cr = i / NT, j = i % NT;
      int p = n0 + j;
      int d = p / (DOUT * DOUT), hh = (p / DOUT) % DOUT, ww = p % DOUT;
      int iz = d + dz, iy = hh + dy, ix = ww + dx;
      __bf16 v = f2bf(0.f);
      if ((unsigned)iz < (unsigned)DOUT && (unsigned)iy < (unsigned)DOUT &&
          (unsigned)ix < (unsigned)DOUT)
        v = ab[(size_t)(cr + ch) * NPOS + (iz * DOUT + iy) * DOUT + ix];
      Bt[buf][j][cr] = v;
    }
  };

  v8f acc[4] = {};

  stageA(0, 0);
  stageB(0, 0);
  for (int ks = 0; ks < 54; ++ks) {        // 27 taps x 2 channel-halves
    const int buf = ks & 1;
    __syncthreads();
    if (ks + 1 < 54) { stageA(ks + 1, buf ^ 1); stageB(ks + 1, buf ^ 1); }
    v16bf bf = load_bfrag(&Bt[buf][nsub + col][0], kg);
#pragma unroll
    for (int mt = 0; mt < 4; ++mt) {
      v16bf af = load_afrag(&At[buf][mt * 16 + col][0], kg);
      acc[mt] = __builtin_amdgcn_wmma_f32_16x16x32_bf16(
          false, af, false, bf, (short)0, acc[mt], false, false);
    }
  }

  float* y2b = y2 + (size_t)b * COUT * NPOS;
  const int n = n0 + nsub + col;
#pragma unroll
  for (int mt = 0; mt < 4; ++mt)
#pragma unroll
    for (int r = 0; r < 8; ++r) {
      int m = mt * 16 + r + 8 * kg;
      y2b[(size_t)m * NPOS + n] = acc[mt][r];
    }
}

// ---------------- per-(b,c) instance-norm statistics ----------------
__global__ __launch_bounds__(256)
void k_stats(const float* __restrict__ y, float* __restrict__ mean,
             float* __restrict__ rstd) {
  const int c = blockIdx.x;                 // flattened b*COUT + c
  const float* p = y + (size_t)c * NPOS;
  float s = 0.f, q = 0.f;
  for (int i = threadIdx.x; i < NPOS; i += 256) {
    float v = p[i];
    s += v; q += v * v;
  }
  __shared__ float ss[8], qq[8];
  int lane = threadIdx.x & 31, wid = threadIdx.x >> 5;
  for (int off = 16; off > 0; off >>= 1) {
    s += __shfl_down(s, off, 32);
    q += __shfl_down(q, off, 32);
  }
  if (lane == 0) { ss[wid] = s; qq[wid] = q; }
  __syncthreads();
  if (threadIdx.x == 0) {
    float S = 0.f, Q = 0.f;
    for (int i = 0; i < 8; ++i) { S += ss[i]; Q += qq[i]; }
    float m = S / (float)NPOS;
    float v = Q / (float)NPOS - m * m;
    mean[c] = m;
    rstd[c] = rsqrtf(v + EPS_F);
  }
}

// ---------------- norm + LeakyReLU -> bf16 (conv2 input) ----------------
__global__ __launch_bounds__(256)
void k_norm_lrelu_bf16(const float* __restrict__ y, const float* __restrict__ mean,
                       const float* __restrict__ rstd, __bf16* __restrict__ out,
                       int n) {
  int idx = blockIdx.x * 256 + threadIdx.x;
  if (idx >= n) return;
  int c = idx / NPOS;
  float v = (y[idx] - mean[c]) * rstd[c];
  v = v > 0.f ? v : 0.01f * v;
  out[idx] = f2bf(v);
}

// ---------------- final: lrelu(norm(y2) + norm(y3)) ----------------
__global__ __launch_bounds__(256)
void k_final(const float* __restrict__ y2, const float* __restrict__ y3,
             const float* __restrict__ m2, const float* __restrict__ r2,
             const float* __restrict__ m3, const float* __restrict__ r3,
             float* __restrict__ out, int n) {
  int idx = blockIdx.x * 256 + threadIdx.x;
  if (idx >= n) return;
  int c = idx / NPOS;
  float v = (y2[idx] - m2[c]) * r2[c] + (y3[idx] - m3[c]) * r3[c];
  out[idx] = v > 0.f ? v : 0.01f * v;
}

extern "C" void kernel_launch(void* const* d_in, const int* in_sizes, int n_in,
                              void* d_out, int out_size, void* d_ws, size_t ws_size,
                              hipStream_t stream) {
  (void)in_sizes; (void)n_in; (void)out_size; (void)ws_size;
  const float* x      = (const float*)d_in[0];   // [2,32,48,48,48]
  const float* prompt = (const float*)d_in[1];   // [2,512]
  const float* rw     = (const float*)d_in[2];   // [4,512]
  const float* rb     = (const float*)d_in[3];   // [4]
  const float* w1     = (const float*)d_in[4];   // [4,64,32,3,3,3]
  // d_in[5] conv1_b: folded away by instance norm
  const float* w2     = (const float*)d_in[6];   // [4,64,64,3,3,3]
  // d_in[7] conv2_b: folded away
  const float* w3     = (const float*)d_in[8];   // [4,64,32,1,1,1]
  // d_in[9] conv3_b folded away; d_in[10] tau unused in eval
  float* out = (float*)d_out;                    // [2,64,24,24,24] fp32

  const size_t nel = (size_t)2 * COUT * NPOS;    // 1,769,472

  char* ws = (char*)d_ws;
  int*    sel = (int*)ws;                               // 2 ints (256B reserved)
  float*  y1  = (float*)(ws + 256);                     // fp32 [2,64,13824]
  float*  y3  = y1 + nel;
  float*  y2  = y3 + nel;
  __bf16* a1  = (__bf16*)(y2 + nel);                    // bf16 [2,64,13824]
  float*  st  = (float*)((char*)a1 + nel * sizeof(__bf16));
  float *m1 = st, *r1 = st + 128, *m3 = st + 256, *r3 = st + 384,
        *m2 = st + 512, *r2 = st + 640;
  __bf16* w1t = (__bf16*)((char*)st + 4096);            // [4][27][64][32] bf16
  __bf16* w2t = w1t + (size_t)4 * 27 * 64 * 32;         // [4][54][64][32] bf16

  // independent prep kernels first
  k_prep_w1<<<(4 * 27 * 64 * 32 + 255) / 256, 256, 0, stream>>>(w1, w1t);
  k_prep_w2<<<(4 * 54 * 64 * 32 + 255) / 256, 256, 0, stream>>>(w2, w2t);
  k_router<<<1, 256, 0, stream>>>(prompt, rw, rb, sel);

  dim3 gconv(NPOS / NT, 2);                             // 108 x 2 blocks
  k_conv1_conv3<<<gconv, 256, 0, stream>>>(x, w1t, w3, sel, y1, y3);

  k_stats<<<2 * COUT, 256, 0, stream>>>(y1, m1, r1);
  k_stats<<<2 * COUT, 256, 0, stream>>>(y3, m3, r3);

  int nblk = (int)((nel + 255) / 256);
  k_norm_lrelu_bf16<<<nblk, 256, 0, stream>>>(y1, m1, r1, a1, (int)nel);

  k_conv2<<<gconv, 256, 0, stream>>>(a1, w2t, sel, y2);

  k_stats<<<2 * COUT, 256, 0, stream>>>(y2, m2, r2);

  k_final<<<nblk, 256, 0, stream>>>(y2, y3, m2, r2, m3, r3, out, (int)nel);
}
